// ContrastiveLoss_24739011625310
// MI455X (gfx1250) — matrix operand order repaired and checked by reading, not compiled, not run on hardware
//
#include <hip/hip_runtime.h>
#include <hip/hip_bf16.h>

// ---------------------------------------------------------------------------
// ContrastiveLoss on MI455X (gfx1250): two 6272x6272x768 cosine GEMMs fused
// with max-over-patch reduction + margin losses, using v_wmma_f32_16x16x32_f16.
// ---------------------------------------------------------------------------

typedef __attribute__((ext_vector_type(16))) _Float16 v16h;
typedef __attribute__((ext_vector_type(8)))  float    v8f;
typedef __attribute__((ext_vector_type(4)))  _Float16 v4h;

#define NIMG   32
#define PP     196
#define DD     768
#define NROWS  (NIMG * PP)          // 6272
#define NN_PAIRS 496                // 32*31/2
#define ND_PAIRS 1024
#define TOTAL_PAIRS (NN_PAIRS + ND_PAIRS)   // 1520
#define QBLK   4                    // 4 x 64-column q-blocks covers 196 (padded to 256)
#define NBLKS  (TOTAL_PAIRS * QBLK) // 6080
#define KC     64                   // K chunk (f32 elements) staged in LDS per iter
#define MT_N   13                   // 13 row tiles of 16 (196 -> 208)
#define CT_N   4                    // 4 col tiles of 16 per block (64 cols)
#define TILES  (MT_N * CT_N)        // 52
#define NEG_INF (-3.0e38f)

#define POS_SCALE (1.0f / (496.0f * 196.0f))
#define NEG_SCALE (1.0f / (1024.0f * 196.0f))

// ---------------------------------------------------------------------------
// Kernel 1: per-row inverse norms  inv = 1/(||x|| + 1e-8).  One wave per row.
// ---------------------------------------------------------------------------
__global__ __launch_bounds__(256) void cl_norms_kernel(
    const float* __restrict__ nrm, const float* __restrict__ dfc,
    float* __restrict__ invN, float* __restrict__ invD)
{
    const int gwave = (int)((blockIdx.x * blockDim.x + threadIdx.x) >> 5);
    const int lane  = (int)(threadIdx.x & 31);
    if (gwave >= 2 * NROWS) return;   // uniform per wave

    const float* src;
    float* dst;
    int row;
    if (gwave < NROWS) { src = nrm; dst = invN; row = gwave; }
    else               { src = dfc; dst = invD; row = gwave - NROWS; }

    const float4* p = (const float4*)(src + (size_t)row * DD);
    float ss = 0.0f;
    for (int c4 = lane; c4 < DD / 4; c4 += 32) {
        float4 v = p[c4];
        ss += v.x * v.x + v.y * v.y + v.z * v.z + v.w * v.w;
    }
#pragma unroll
    for (int off = 16; off >= 1; off >>= 1)
        ss += __shfl_xor(ss, off, 32);

    if (lane == 0) dst[row] = 1.0f / (sqrtf(ss) + 1e-8f);
}

// ---------------------------------------------------------------------------
// Kernel 2: per (pair, q-block) fused GEMM + max-over-p + margin-loss partial.
//   A tile: 196(+12 pad) rows of image i (normal), B tile: 64 q-rows of image j.
//   S = A_f16 . B_f16^T with f32 WMMA accumulation.
// ---------------------------------------------------------------------------
__global__ __launch_bounds__(256) void cl_gemm_kernel(
    const float* __restrict__ nrm, const float* __restrict__ dfc,
    const float* __restrict__ invN, const float* __restrict__ invD,
    float* __restrict__ partials)
{
    __shared__ _Float16 As[208][72];   // row-major [p][k], stride 144B (16B mult)
    __shared__ _Float16 Bs[64][72];    // row-major [q][k]
    __shared__ float    red[8][CT_N][16];

    const int bid  = (int)blockIdx.x;
    const int qb   = bid & (QBLK - 1);
    const int pair = bid >> 2;
    const bool is_nn = pair < NN_PAIRS;

    int i, j;
    if (is_nn) {
        int t = pair, ii = 0;
        while (t >= 31 - ii) { t -= 31 - ii; ++ii; }
        i = ii; j = ii + 1 + t;
    } else {
        int t = pair - NN_PAIRS;
        i = t >> 5; j = t & 31;
    }

    const float* Abase = nrm + (size_t)i * PP * DD;
    const float* Bbase = (is_nn ? nrm : dfc) + (size_t)j * PP * DD;
    const float* invA  = invN + i * PP;
    const float* invB  = (is_nn ? invN : invD) + j * PP;

    const int tid  = (int)threadIdx.x;
    const int lane = tid & 31;
    // Wave id is uniform across the wave: pin it into an SGPR so the per-wave
    // tile guards become scalar branches (EXEC stays all-ones around WMMA, as
    // the ISA requires), instead of s_and_saveexec predication.
    const int wave = __builtin_amdgcn_readfirstlane(tid >> 5);
    const int half = lane >> 4;
    const int lrow = lane & 15;

    v8f acc[7];
    const v8f vzero = {0.f, 0.f, 0.f, 0.f, 0.f, 0.f, 0.f, 0.f};
#pragma unroll
    for (int s = 0; s < 7; ++s) acc[s] = vzero;

    for (int k0 = 0; k0 < DD; k0 += KC) {
        // ---- stage A chunk: 208 rows x 64 k (normalize f32 -> f16) ----
        for (int idx = tid; idx < 208 * (KC / 4); idx += 256) {
            int r  = idx >> 4;
            int c4 = (idx & 15) << 2;          // f32 column within chunk
            float4 v = make_float4(0.f, 0.f, 0.f, 0.f);
            float  sc = 0.0f;
            if (r < PP) {
                v  = *(const float4*)(Abase + (size_t)r * DD + k0 + c4);
                sc = invA[r];
            }
            v4h h;
            h.x = (_Float16)(v.x * sc);
            h.y = (_Float16)(v.y * sc);
            h.z = (_Float16)(v.z * sc);
            h.w = (_Float16)(v.w * sc);
            *(v4h*)(&As[r][c4]) = h;
        }
        // ---- stage B chunk: 64 q-rows x 64 k ----
        for (int idx = tid; idx < 64 * (KC / 4); idx += 256) {
            int r  = idx >> 4;
            int c4 = (idx & 15) << 2;
            int q  = qb * 64 + r;
            float4 v = make_float4(0.f, 0.f, 0.f, 0.f);
            float  sc = 0.0f;
            if (q < PP) {
                v  = *(const float4*)(Bbase + (size_t)q * DD + k0 + c4);
                sc = invB[q];
            }
            v4h h;
            h.x = (_Float16)(v.x * sc);
            h.y = (_Float16)(v.y * sc);
            h.z = (_Float16)(v.z * sc);
            h.w = (_Float16)(v.w * sc);
            *(v4h*)(&Bs[r][c4]) = h;
        }
        __syncthreads();

        // ---- WMMA over this K chunk ----
#pragma unroll
        for (int s = 0; s < 7; ++s) {
            const int tile = wave + (s << 3);   // scalar (wave-uniform)
            if (tile < TILES) {
                const int mt = tile % MT_N;
                const int ct = tile / MT_N;
                const int arow = mt * 16 + lrow;   // A: lane half selects K sub-range
                const int bcol = ct * 16 + lrow;   // B: lane half selects K sub-range
#pragma unroll
                for (int ko = 0; ko < KC; ko += 32) {
                    // A fragment (16x32 f16): VGPR0-3 K=half*8+0..7, VGPR4-7 K=16+half*8+0..7
                    union { v16h v; float4 f[2]; } af;
                    af.f[0] = *(const float4*)(&As[arow][ko + half * 8]);
                    af.f[1] = *(const float4*)(&As[arow][ko + 16 + half * 8]);
                    // B fragment (32x16 f16): VGPR0-3 K=half*8+0..7, VGPR4-7 K=16+half*8+0..7
                    union { v16h v; float4 f[2]; } bf;
                    bf.f[0] = *(const float4*)(&Bs[bcol][ko + half * 8]);
                    bf.f[1] = *(const float4*)(&Bs[bcol][ko + 16 + half * 8]);

                    acc[s] = __builtin_amdgcn_wmma_f32_16x16x32_f16(
                        false, af.v, false, bf.v, (short)0, acc[s], false, false);
                }
            }
        }
        __syncthreads();
    }

    // ---- max over p (rows of S) per column ----
    float wmax[CT_N];
#pragma unroll
    for (int c = 0; c < CT_N; ++c) wmax[c] = NEG_INF;

#pragma unroll
    for (int s = 0; s < 7; ++s) {
        const int tile = wave + (s << 3);       // scalar (wave-uniform)
        if (tile < TILES) {
            const int mt = tile % MT_N;
            const int ct = tile / MT_N;
            float m;
            if (mt < MT_N - 1) {
                // all 16 rows valid: max over 8 accumulator VGPRs
                m = fmaxf(fmaxf(fmaxf(acc[s][0], acc[s][1]), fmaxf(acc[s][2], acc[s][3])),
                          fmaxf(fmaxf(acc[s][4], acc[s][5]), fmaxf(acc[s][6], acc[s][7])));
            } else {
                // rows 192..195 valid only: M=0..3 -> VGPR0..3, low lane-half only
                m = (half == 0)
                        ? fmaxf(fmaxf(acc[s][0], acc[s][1]), fmaxf(acc[s][2], acc[s][3]))
                        : NEG_INF;
            }
            // combine lane halves (lane L and L+16 hold the same column)
            m = fmaxf(m, __shfl_xor(m, 16, 32));
#pragma unroll
            for (int c = 0; c < CT_N; ++c)
                if (c == ct) wmax[c] = fmaxf(wmax[c], m);
        }
    }

    if (lane < 16) {
#pragma unroll
        for (int c = 0; c < CT_N; ++c) red[wave][c][lrow] = wmax[c];
    }
    __syncthreads();

    if (wave == 0) {
        float contrib = 0.0f;
        if (lane < 16) {
#pragma unroll
            for (int c = 0; c < CT_N; ++c) {
                float cm = red[0][c][lrow];
#pragma unroll
                for (int w = 1; w < 8; ++w) cm = fmaxf(cm, red[w][c][lrow]);
                const int q = qb * 64 + c * 16 + lrow;
                if (q < PP) {
                    contrib += is_nn ? (1.0f - cm) * POS_SCALE
                                     : fmaxf(cm - 0.5f, 0.0f) * NEG_SCALE;
                }
            }
        }
#pragma unroll
        for (int off = 16; off >= 1; off >>= 1)
            contrib += __shfl_xor(contrib, off, 32);
        if (lane == 0) partials[bid] = contrib;
    }
}

// ---------------------------------------------------------------------------
// Kernel 3: deterministic fixed-order sum of partials -> d_out[0]
// ---------------------------------------------------------------------------
__global__ __launch_bounds__(256) void cl_finalize_kernel(
    const float* __restrict__ partials, float* __restrict__ out)
{
    __shared__ float sm[256];
    float s = 0.0f;
    for (int idx = (int)threadIdx.x; idx < NBLKS; idx += 256) s += partials[idx];
    sm[threadIdx.x] = s;
    __syncthreads();
#pragma unroll
    for (int off = 128; off > 0; off >>= 1) {
        if ((int)threadIdx.x < off) sm[threadIdx.x] += sm[threadIdx.x + off];
        __syncthreads();
    }
    if (threadIdx.x == 0) out[0] = sm[0];
}

// ---------------------------------------------------------------------------
extern "C" void kernel_launch(void* const* d_in, const int* in_sizes, int n_in,
                              void* d_out, int out_size, void* d_ws, size_t ws_size,
                              hipStream_t stream) {
    (void)in_sizes; (void)n_in; (void)out_size; (void)ws_size;
    const float* nrm = (const float*)d_in[0];   // [32,196,768] f32
    const float* dfc = (const float*)d_in[1];   // [32,196,768] f32
    float* invN     = (float*)d_ws;             // 6272 f32
    float* invD     = invN + NROWS;             // 6272 f32
    float* partials = invD + NROWS;             // 6080 f32
    float* out      = (float*)d_out;            // 1 f32

    // 1 wave per row, 8 waves per block -> (2*6272)/8 = 1568 blocks
    cl_norms_kernel<<<(2 * NROWS) / 8, 256, 0, stream>>>(nrm, dfc, invN, invD);
    cl_gemm_kernel<<<NBLKS, 256, 0, stream>>>(nrm, dfc, invN, invD, partials);
    cl_finalize_kernel<<<1, 256, 0, stream>>>(partials, out);
}
